// UniversalMoEContainer_26310969655839
// MI455X (gfx1250) — matrix operand order repaired and checked by reading, not compiled, not run on hardware
//
#include <hip/hip_runtime.h>

typedef __attribute__((ext_vector_type(16))) _Float16 v16h;
typedef __attribute__((ext_vector_type(8)))  _Float16 v8h;
typedef __attribute__((ext_vector_type(8)))  float    v8f;
typedef __attribute__((ext_vector_type(4)))  float    v4f;

#define C_IN   128
#define C_OUT  128
#define C_HID  256
#define HW     1600
#define NTILE  80
#define NT_PER_B (HW / NTILE)     // 20 spatial tiles per batch image
#define XT_PITCH 136              // halves per Xt row (128 + 8 pad, 272 B, 16B-aligned)
#define HT_PITCH 264              // halves per Ht row (256 + 8 pad, 528 B, 16B-aligned)
#define W1_ELEMS (8 * C_HID * C_IN)   // 262144
#define W2_ELEMS (8 * C_OUT * C_HID)  // 262144

static __device__ __forceinline__ v16h cat8(v8h lo, v8h hi) {
    return __builtin_shufflevector(lo, hi, 0,1,2,3,4,5,6,7,8,9,10,11,12,13,14,15);
}

// ---- prep: one-time fp32 -> f16 conversion of expert weights into workspace ----
__global__ __launch_bounds__(256)
void convert_weights_kernel(const float* __restrict__ W1,
                            const float* __restrict__ W2,
                            _Float16* __restrict__ wsh)
{
    const int i = blockIdx.x * 256 + threadIdx.x;
    if (i < W1_ELEMS) {
        wsh[i]            = (_Float16)W1[i];
        wsh[W1_ELEMS + i] = (_Float16)W2[i];
    }
}

__global__ __launch_bounds__(256)
void moe_1x1_wmma_kernel(const float* __restrict__ x,      // [B,128,1600]
                         const float* __restrict__ gatew,  // [B,2]
                         const int*   __restrict__ gidx,   // [B,2]
                         const _Float16* __restrict__ W1h, // [8,256,128] f16
                         const float* __restrict__ b1,     // [8,256]
                         const _Float16* __restrict__ W2h, // [8,128,256] f16
                         const float* __restrict__ b2,     // [8,128]
                         float* __restrict__ out)          // [B,128,1600]
{
    __shared__ __align__(16) _Float16 Xt[NTILE * XT_PITCH]; // X tile, transposed: [n][c]
    __shared__ __align__(16) _Float16 Ht[NTILE * HT_PITCH]; // hidden tile, transposed: [n][m]

    const int blk  = blockIdx.x;
    const int b    = blk / NT_PER_B;
    const int t    = blk % NT_PER_B;
    const int tid  = threadIdx.x;
    const int lane = tid & 31;
    const int wave = tid >> 5;
    const int g    = lane >> 4;   // half-wave group (K-split in WMMA layouts)
    const int l16  = lane & 15;

    // ---- stage X tile: fp32 global (float4, coalesced) -> f16 LDS, transposed ----
    const float* xb = x + (size_t)b * (C_IN * HW) + t * NTILE;
    for (int idx = tid; idx < C_IN * (NTILE / 4); idx += 256) {
        const int c  = idx / (NTILE / 4);
        const int n4 = idx % (NTILE / 4);
        const v4f v = *(const v4f*)(xb + (size_t)c * HW + n4 * 4);
        Xt[(n4 * 4 + 0) * XT_PITCH + c] = (_Float16)v.x;
        Xt[(n4 * 4 + 1) * XT_PITCH + c] = (_Float16)v.y;
        Xt[(n4 * 4 + 2) * XT_PITCH + c] = (_Float16)v.z;
        Xt[(n4 * 4 + 3) * XT_PITCH + c] = (_Float16)v.w;
    }
    __syncthreads();

    // persistent fp32 output accumulators: wave owns M-tile = wave, 5 N-tiles
    v8f acc2[5];
    #pragma unroll
    for (int i = 0; i < 5; ++i) acc2[i] = v8f{};

    const float w0  = gatew[b * 2 + 0];
    const float w1g = gatew[b * 2 + 1];
    const int   e0  = gidx[b * 2 + 0];
    const int   e1  = gidx[b * 2 + 1];

    for (int slot = 0; slot < 2; ++slot) {
        const int   e = slot ? e1 : e0;
        const float w = slot ? w1g : w0;
        const _Float16* W1e = W1h + (size_t)e * (C_HID * C_IN);
        const float*    b1e = b1  + (size_t)e * C_HID;
        const _Float16* W2e = W2h + (size_t)e * (C_OUT * C_HID);
        const float*    b2e = b2  + (size_t)e * C_OUT;

        // ---- GEMM1: H(256 x 80) = silu(W1e(256x128) * Xt + b1) ----
        // wave owns mt = wave*2 + j; A frags + bias loaded once, reused over 5 N-tiles
        #pragma unroll
        for (int j = 0; j < 2; ++j) {
            const int mt = wave * 2 + j;
            const _Float16* arow = W1e + (size_t)(mt * 16 + l16) * C_IN;  // A: M = lane&15
            v16h afr[4];
            #pragma unroll
            for (int kc = 0; kc < 4; ++kc)
                afr[kc] = cat8(*(const v8h*)(arow + kc * 32 + g * 8),
                               *(const v8h*)(arow + kc * 32 + 16 + g * 8));
            float bias[8];
            #pragma unroll
            for (int r = 0; r < 8; ++r) bias[r] = b1e[mt * 16 + 8 * g + r];

            for (int nt = 0; nt < 5; ++nt) {
                const _Float16* brow = &Xt[(nt * 16 + l16) * XT_PITCH];   // B: N = lane&15
                v8f acc = v8f{};
                #pragma unroll
                for (int kc = 0; kc < 4; ++kc) {
                    const v16h bf = cat8(*(const v8h*)(brow + kc * 32 + g * 8),
                                         *(const v8h*)(brow + kc * 32 + 16 + g * 8));
                    acc = __builtin_amdgcn_wmma_f32_16x16x32_f16(false, afr[kc], false, bf,
                                                                 (short)0, acc, false, false);
                }
                // bias + SiLU (fast v_rcp), pack f16, store transposed (VGPR r -> M = r + 8g)
                v8h hq;
                #pragma unroll
                for (int r = 0; r < 8; ++r) {
                    const float v = acc[r] + bias[r];
                    hq[r] = (_Float16)(v * __builtin_amdgcn_rcpf(1.0f + __expf(-v)));
                }
                *(v8h*)&Ht[(nt * 16 + l16) * HT_PITCH + mt * 16 + 8 * g] = hq;
            }
        }
        __syncthreads();

        // ---- GEMM2: Y(128 x 80) = W2e(128x256) * H ; wave owns mt = wave ----
        {
            const _Float16* arow = W2e + (size_t)(wave * 16 + l16) * C_HID;
            v16h afr[8];
            #pragma unroll
            for (int kc = 0; kc < 8; ++kc)
                afr[kc] = cat8(*(const v8h*)(arow + kc * 32 + g * 8),
                               *(const v8h*)(arow + kc * 32 + 16 + g * 8));
            float bias[8];
            #pragma unroll
            for (int r = 0; r < 8; ++r) bias[r] = b2e[wave * 16 + 8 * g + r];

            for (int nt = 0; nt < 5; ++nt) {
                const _Float16* brow = &Ht[(nt * 16 + l16) * HT_PITCH];
                v8f acc = v8f{};
                #pragma unroll
                for (int kc = 0; kc < 8; ++kc) {
                    const v16h bf = cat8(*(const v8h*)(brow + kc * 32 + g * 8),
                                         *(const v8h*)(brow + kc * 32 + 16 + g * 8));
                    acc = __builtin_amdgcn_wmma_f32_16x16x32_f16(false, afr[kc], false, bf,
                                                                 (short)0, acc, false, false);
                }
                #pragma unroll
                for (int r = 0; r < 8; ++r)
                    acc2[nt][r] += w * (acc[r] + bias[r]);   // gate * (y + b2), per expert slot
            }
        }
        __syncthreads();   // Ht reused by next expert slot
    }

    // ---- write output (each block owns its [b, :, tile] slab exclusively) ----
    float* ob = out + (size_t)b * (C_OUT * HW) + t * NTILE;
    for (int nt = 0; nt < 5; ++nt) {
        const int n = nt * 16 + l16;
        #pragma unroll
        for (int r = 0; r < 8; ++r) {
            const int mr = wave * 16 + 8 * g + r;
            ob[(size_t)mr * HW + n] = acc2[nt][r];
        }
    }
}

extern "C" void kernel_launch(void* const* d_in, const int* in_sizes, int n_in,
                              void* d_out, int out_size, void* d_ws, size_t ws_size,
                              hipStream_t stream) {
    const float* x     = (const float*)d_in[0];
    const float* gatew = (const float*)d_in[1];
    const int*   gidx  = (const int*)  d_in[2];
    const float* W1    = (const float*)d_in[3];
    const float* b1    = (const float*)d_in[4];
    const float* W2    = (const float*)d_in[5];
    const float* b2    = (const float*)d_in[6];
    float* out = (float*)d_out;

    _Float16* wsh = (_Float16*)d_ws;              // [W1h | W2h] = 1 MB f16
    const _Float16* W1h = wsh;
    const _Float16* W2h = wsh + W1_ELEMS;

    convert_weights_kernel<<<(W1_ELEMS + 255) / 256, 256, 0, stream>>>(W1, W2, wsh);

    const int blocks = 128 * NT_PER_B;            // B * spatial tiles = 2560
    moe_1x1_wmma_kernel<<<blocks, 256, 0, stream>>>(x, gatew, gidx, W1h, b1, W2h, b2, out);
}
